// Intermediate_63513976373691
// MI455X (gfx1250) — matrix operand order repaired and checked by reading, not compiled, not run on hardware
//
#include <hip/hip_runtime.h>

// ---------------- problem dims ----------------
#define B_    2048
#define T_    16
#define D_    128
#define H_    8
#define HD_   1024
#define DFF_  512
#define FFIN_ 1152      // (H+1)*D
#define DOUT_ 129       // D+1
#define BT_   16        // batch rows per workgroup (== WMMA M tile)
#define NTHR  256
#define NWAVE 8

// ---------------- LDS layout (byte offsets) ----------------
#define OFF_XLN    0        // ushort [256][128]    65536  (LN output, rows r=b*16+t)
#define OFF_V1     65536    // ushort [16][16][128] 65536  (per-stream v1)
#define OFF_QH     131072   // ushort [16][8][128]  32768  (per-head q)
#define OFF_KHT    163840   // ushort [16][128][16] 65536  (per-head k, d-major for B-frags)
#define OFF_WPROB  229376   // float  [16][16][16]  16384  (softmax / w2 scratch)
#define OFF_WSTG   245760   // ushort [8][512]      8192   (per-wave 32x16 weight staging)
#define SMEM_BYTES 253952
// FFN-phase reuse of dead regions:
#define OFF_XAT    0        // ushort [16][1152]  (in XLN)    <- TDM load target
#define OFF_MID    65536    // float  [16][1024]  (in V1)
#define OFF_HB     131072   // ushort [16][512]   (in QH)
#define OFF_OUTB   131072   // float  [16][128]   (in QH, after HB dead) <- TDM store source
#define OFF_FFOUT  229376   // float  [16][144]   (in WPROB)

// ---------------- d_ws layout (ushort element offsets) ----------------
#define WSE_XAG     0                               // bf16 [B][T][1152]  37,748,736
#define WSE_QW0     37748736                        // bf16 8*1024*128
#define WSE_QW1     (WSE_QW0 + 1048576)
#define WSE_KAW0    (WSE_QW1 + 1048576)
#define WSE_KAW1    (WSE_KAW0 + 131072)
#define WSE_KBW0    (WSE_KAW1 + 131072)
#define WSE_KBW1    (WSE_KBW0 + 131072)
#define WSE_V1AW0   (WSE_KBW1 + 131072)
#define WSE_V1AW1   (WSE_V1AW0 + 16384)
#define WSE_V1BW0   (WSE_V1AW1 + 16384)
#define WSE_V1BW1   (WSE_V1BW0 + 16384)
#define WSE_FFINW   (WSE_V1BW1 + 16384)             // bf16 16*1024*1152
#define WSE_FFOUTW  (WSE_FFINW + 18874368)          // bf16 16*129*512

typedef __attribute__((ext_vector_type(16))) __bf16       v16bf;
typedef __attribute__((ext_vector_type(8)))  float        v8f;
typedef __attribute__((ext_vector_type(4)))  unsigned int v4u;
typedef __attribute__((ext_vector_type(4)))  unsigned int tdm_u32x4;
typedef __attribute__((ext_vector_type(8)))  int          tdm_i32x8;
typedef __attribute__((ext_vector_type(4)))  int          tdm_i32x4;

union ABFrag { v16bf v; v4u q[2]; };

__device__ __forceinline__ unsigned short f2bf(float f) {
    unsigned u = __builtin_bit_cast(unsigned, f);
    unsigned r = u + 0x7FFFu + ((u >> 16) & 1u);   // round-to-nearest-even
    return (unsigned short)(r >> 16);
}
__device__ __forceinline__ float bf2f(unsigned short h) {
    unsigned u = ((unsigned)h) << 16;
    return __builtin_bit_cast(float, u);
}
__device__ __forceinline__ v8f zero8() {
    v8f z;
#pragma unroll
    for (int i = 0; i < 8; ++i) z[i] = 0.f;
    return z;
}
__device__ __forceinline__ v8f wmma_bf16(v16bf a, v16bf b, v8f c) {
    return __builtin_amdgcn_wmma_f32_16x16x32_bf16(false, a, false, b, (short)0, c,
                                                   false, false);
}

// A-frag (16x32 bf16): row-major source As[m][k]
__device__ __forceinline__ v16bf load_a_frag(const unsigned short* As, int row_stride,
                                             int kbase) {
    int lane = threadIdx.x & 31;
    int m = lane & 15, half = lane >> 4;
    const v4u* p = (const v4u*)(As + m * row_stride + kbase + half * 8);
    ABFrag r;
    r.q[0] = p[0];
    r.q[1] = p[2];
    return r.v;
}
// Same, rows m>=8 read as zero (M padded 8->16)
__device__ __forceinline__ v16bf load_a_frag_m8(const unsigned short* As, int row_stride,
                                                int kbase) {
    int lane = threadIdx.x & 31;
    int m = lane & 15, half = lane >> 4;
    bool valid = m < 8;
    const v4u* p = (const v4u*)(As + (valid ? m : 0) * row_stride + kbase + half * 8);
    v4u z; z[0] = 0u; z[1] = 0u; z[2] = 0u; z[3] = 0u;
    ABFrag r;
    r.q[0] = valid ? p[0] : z;
    r.q[1] = valid ? p[2] : z;
    return r.v;
}
// B-frag (32x16 bf16): k-major source Bs[k][16]
__device__ __forceinline__ v16bf load_b_frag(const unsigned short* Bs) {
    int lane = threadIdx.x & 31;
    int n = lane & 15, half = lane >> 4;
    v16bf r;
#pragma unroll
    for (int s = 0; s < 16; ++s) {
        int k = (s < 8) ? (half * 8 + s) : (16 + half * 8 + (s - 8));
        r[s] = __builtin_bit_cast(__bf16, Bs[k * 16 + n]);
    }
    return r;
}
// Stage 32(k)x16(n) k-major tile from PRE-CONVERTED bf16 row-major W[n][k]
__device__ __forceinline__ void stage_b_tile_bf(unsigned short* stg,
                                                const unsigned short* W, int ld,
                                                int nbase, int kbase, int nmax) {
    int lane = threadIdx.x & 31;
#pragma unroll
    for (int i = 0; i < 16; ++i) {
        int idx = i * 32 + lane;
        int n = idx & 15;
        int k = idx >> 4;
        unsigned short v =
            (nbase + n < nmax) ? W[(size_t)(nbase + n) * ld + kbase + k] : (unsigned short)0;
        stg[k * 16 + n] = v;
    }
}

// ---- TDM helpers: issue a 2D tile move between global and LDS ----
// clang-23 / therock-10.0 builtin arity: (u32x4, i32x8, i32x4, i32x4, i32x8, i32 cpol)
__device__ __forceinline__ void tdm_load_2d(unsigned long long gaddr, unsigned lds_byte,
                                            int elems_per_row, int rows,
                                            int row_stride_elems, int data_size_code) {
    tdm_u32x4 g0; tdm_i32x8 g1; tdm_i32x4 g2, g3; tdm_i32x8 g4;
    g0[0] = 1u;                                             // count=1
    g0[1] = lds_byte;                                       // LDS byte address
    g0[2] = (unsigned)(gaddr & 0xFFFFFFFFull);              // global addr [31:0]
    g0[3] = (unsigned)((gaddr >> 32) & 0x01FFFFFFull) | (2u << 30);  // [56:32] | type=2
    g1[0] = (data_size_code << 16);                         // data_size, mask=0
    g1[1] = (int)((unsigned)elems_per_row << 16);           // tensor_dim0[15:0]
    g1[2] = (int)((unsigned)rows << 16) |
            (int)(((unsigned)elems_per_row >> 16) & 0xFFFF);// dim1[15:0] | dim0[31:16]
    g1[3] = (int)((unsigned)elems_per_row << 16);           // tile_dim0 | dim1[31:16]=0
    g1[4] = rows;                                           // tile_dim1 (tile_dim2=0)
    g1[5] = row_stride_elems;                               // tensor_dim0_stride[31:0]
    g1[6] = 0; g1[7] = 0;
#pragma unroll
    for (int i = 0; i < 4; ++i) { g2[i] = 0; g3[i] = 0; }
#pragma unroll
    for (int i = 0; i < 8; ++i) g4[i] = 0;
    __builtin_amdgcn_tensor_load_to_lds(g0, g1, g2, g3, g4, 0);
}
__device__ __forceinline__ void tdm_store_2d(unsigned long long gaddr, unsigned lds_byte,
                                             int elems_per_row, int rows,
                                             int row_stride_elems, int data_size_code) {
    tdm_u32x4 g0; tdm_i32x8 g1; tdm_i32x4 g2, g3; tdm_i32x8 g4;
    g0[0] = 1u;
    g0[1] = lds_byte;
    g0[2] = (unsigned)(gaddr & 0xFFFFFFFFull);
    g0[3] = (unsigned)((gaddr >> 32) & 0x01FFFFFFull) | (2u << 30);
    g1[0] = (data_size_code << 16);
    g1[1] = (int)((unsigned)elems_per_row << 16);
    g1[2] = (int)((unsigned)rows << 16) |
            (int)(((unsigned)elems_per_row >> 16) & 0xFFFF);
    g1[3] = (int)((unsigned)elems_per_row << 16);
    g1[4] = rows;
    g1[5] = row_stride_elems;
    g1[6] = 0; g1[7] = 0;
#pragma unroll
    for (int i = 0; i < 4; ++i) { g2[i] = 0; g3[i] = 0; }
#pragma unroll
    for (int i = 0; i < 8; ++i) g4[i] = 0;
    __builtin_amdgcn_tensor_store_from_lds(g0, g1, g2, g3, g4, 0);
}

struct Params { const float* in[34]; };

// ---------------- bf16 weight pre-conversion (once per launch) ----------------
__global__ __launch_bounds__(NTHR)
void cvt_bf16(const float* __restrict__ src, unsigned short* __restrict__ dst, int n) {
    for (int i = blockIdx.x * blockDim.x + threadIdx.x; i < n; i += gridDim.x * blockDim.x)
        dst[i] = f2bf(src[i]);
}

// d_in dict order:
// 0 x_int 1 x_ext 2 ln_int_g 3 ln_int_b 4 ln_ext_g 5 ln_ext_b 6 hs_int 7 hs_ext
// 8..18  ai_{qw,qb,kaw,kab,kbw,kbb,v1aw,v1ab,v1bw,v1bb,v2w}
// 19..29 ae_{...}
// 30 ff_in_w 31 ff_in_b 32 ff_out_w 33 ff_out_b

__global__ __launch_bounds__(NTHR, 1)
void fused_layer(Params p, unsigned short* ws, float* out) {
    extern __shared__ char smem[];
    unsigned short* XLN   = (unsigned short*)(smem + OFF_XLN);
    unsigned short* V1    = (unsigned short*)(smem + OFF_V1);
    unsigned short* QH    = (unsigned short*)(smem + OFF_QH);
    unsigned short* KHT   = (unsigned short*)(smem + OFF_KHT);
    float*          WPROB = (float*)(smem + OFF_WPROB);
    unsigned short* WSTG  = (unsigned short*)(smem + OFF_WSTG);

    const int tid  = threadIdx.x;
    const int wave = tid >> 5;
    const int lane = tid & 31;
    const int b0   = blockIdx.x * BT_;
    unsigned short* stg = WSTG + wave * 512;
    unsigned short* xag = ws + WSE_XAG;

    // ================= Phase A: LayerNorm -> XLN (bf16) =================
    for (int rr = wave; rr < BT_ * T_; rr += NWAVE) {
        int b = rr >> 4, t = rr & 15;
        const float* src = (t < 8)
            ? (p.in[0] + ((size_t)(b0 + b) * 8 + t) * D_)
            : (p.in[1] + ((size_t)(b0 + b) * 8 + (t - 8)) * D_);
        float4 v = ((const float4*)src)[lane];
        float s = v.x + v.y + v.z + v.w;
#pragma unroll
        for (int m = 16; m >= 1; m >>= 1) s += __shfl_xor(s, m, 32);
        float mean = s * (1.f / 128.f);
        float d0 = v.x - mean, d1 = v.y - mean, d2 = v.z - mean, d3 = v.w - mean;
        float ss = d0 * d0 + d1 * d1 + d2 * d2 + d3 * d3;
#pragma unroll
        for (int m = 16; m >= 1; m >>= 1) ss += __shfl_xor(ss, m, 32);
        float inv = rsqrtf(ss * (1.f / 128.f) + 1e-5f);
        const float* g = (t < 8) ? p.in[2] : p.in[4];
        const float* be = (t < 8) ? p.in[3] : p.in[5];
        int c = lane * 4;
        XLN[rr * 128 + c + 0] = f2bf(d0 * inv * g[c + 0] + be[c + 0]);
        XLN[rr * 128 + c + 1] = f2bf(d1 * inv * g[c + 1] + be[c + 1]);
        XLN[rr * 128 + c + 2] = f2bf(d2 * inv * g[c + 2] + be[c + 2]);
        XLN[rr * 128 + c + 3] = f2bf(d3 * inv * g[c + 3] + be[c + 3]);
    }
    __syncthreads();

    const float inv_sqrt_d = 0.08838834764831845f;  // 1/sqrt(128)

    // ================= Per-stream attention =================
#pragma unroll 1
    for (int s = 0; s < 2; ++s) {
        const int sb = 8 + s * 11;
        const float* qb   = p.in[sb + 1];
        const float* kab  = p.in[sb + 3];
        const float* kbb  = p.in[sb + 5];
        const float* v1ab = p.in[sb + 7];
        const float* v1bb = p.in[sb + 9];
        const float* v2w  = p.in[sb + 10];
        const float* hs   = p.in[6 + s];
        const unsigned short* qwb   = ws + (s ? WSE_QW1 : WSE_QW0);
        const unsigned short* kawb  = ws + (s ? WSE_KAW1 : WSE_KAW0);
        const unsigned short* kbwb  = ws + (s ? WSE_KBW1 : WSE_KBW0);
        const unsigned short* v1awb = ws + (s ? WSE_V1AW1 : WSE_V1AW0);
        const unsigned short* v1bwb = ws + (s ? WSE_V1BW1 : WSE_V1BW0);
        const int abase = s * 8;
        const int bbase = (1 - s) * 8;

        __syncthreads();  // protect V1 rewrite vs previous stream's readers

        // ---- v1[b][t][128] (WMMA, M=16 batch, N=128, K=128) ----
#pragma unroll 1
        for (int t = wave; t < T_; t += NWAVE) {
            const unsigned short* W = (t < 8) ? v1awb : v1bwb;
            const float* bia = (t < 8) ? v1ab : v1bb;
            int src_t = (t < 8) ? (abase + t) : (bbase + (t - 8));
            const unsigned short* Ab = XLN + src_t * 128;
#pragma unroll 1
            for (int nt = 0; nt < 8; ++nt) {
                v8f acc = zero8();
#pragma unroll 1
                for (int kt = 0; kt < 4; ++kt) {
                    stage_b_tile_bf(stg, W, 128, nt * 16, kt * 32, 128);
                    v16bf af = load_a_frag(Ab, 2048, kt * 32);
                    v16bf bf_ = load_b_frag(stg);
                    acc = wmma_bf16(af, bf_, acc);
                }
                int n = lane & 15, half = lane >> 4;
                float bb = bia[nt * 16 + n];
#pragma unroll
                for (int r = 0; r < 8; ++r) {
                    int m = r + half * 8;
                    V1[m * 2048 + t * 128 + nt * 16 + n] = f2bf(acc[r] + bb);
                }
            }
        }

        // ---- per-head loop ----
#pragma unroll 1
        for (int h = 0; h < H_; ++h) {
            __syncthreads();

            // Q projection: wave==a -> QH[b][a][128]
            {
                int a = wave;
                const unsigned short* Wq = qwb + (size_t)a * HD_ * D_;
                const unsigned short* Ab = XLN + (abase + a) * 128;
#pragma unroll 1
                for (int nt = 0; nt < 8; ++nt) {
                    v8f acc = zero8();
#pragma unroll 1
                    for (int kt = 0; kt < 4; ++kt) {
                        stage_b_tile_bf(stg, Wq, 128, h * 128 + nt * 16, kt * 32, HD_);
                        v16bf af = load_a_frag(Ab, 2048, kt * 32);
                        v16bf bf_ = load_b_frag(stg);
                        acc = wmma_bf16(af, bf_, acc);
                    }
                    int n = lane & 15, half = lane >> 4;
                    float bb = qb[(size_t)a * HD_ + h * 128 + nt * 16 + n];
#pragma unroll
                    for (int r = 0; r < 8; ++r) {
                        int m = r + half * 8;
                        QH[m * 1024 + a * 128 + nt * 16 + n] = f2bf(acc[r] + bb);
                    }
                }
            }
            // K projection -> KHT[b][d][t] (d-major)
#pragma unroll 1
            for (int t = wave; t < T_; t += NWAVE) {
                const unsigned short* Wk = (t < 8) ? kawb : kbwb;
                const float* kbi = (t < 8) ? kab : kbb;
                int src = (t < 8) ? (abase + t) : (bbase + (t - 8));
                const unsigned short* Ab = XLN + src * 128;
#pragma unroll 1
                for (int nt = 0; nt < 8; ++nt) {
                    v8f acc = zero8();
#pragma unroll 1
                    for (int kt = 0; kt < 4; ++kt) {
                        stage_b_tile_bf(stg, Wk, 128, h * 128 + nt * 16, kt * 32, HD_);
                        v16bf af = load_a_frag(Ab, 2048, kt * 32);
                        v16bf bf_ = load_b_frag(stg);
                        acc = wmma_bf16(af, bf_, acc);
                    }
                    int n = lane & 15, half = lane >> 4;
                    float bb = kbi[h * 128 + nt * 16 + n];
#pragma unroll
                    for (int r = 0; r < 8; ++r) {
                        int m = r + half * 8;
                        KHT[m * 2048 + (nt * 16 + n) * 16 + t] = f2bf(acc[r] + bb);
                    }
                }
            }
            __syncthreads();

            // scores (WMMA K=128) + softmax + w2 fold + out
#pragma unroll 1
            for (int b = wave; b < BT_; b += NWAVE) {
                const unsigned short* Aq = QH + b * 1024;
                const unsigned short* Bk = KHT + b * 2048;
                v8f sacc = zero8();
#pragma unroll 1
                for (int kt = 0; kt < 4; ++kt) {
                    v16bf af = load_a_frag_m8(Aq, 128, kt * 32);
                    v16bf bf_ = load_b_frag(Bk + kt * 32 * 16);
                    sacc = wmma_bf16(af, bf_, sacc);
                }
                int n = lane & 15, half = lane >> 4;
                float* wp = WPROB + b * 256;
#pragma unroll
                for (int r = 0; r < 8; ++r) {
                    float sv = sacc[r] * inv_sqrt_d;
                    float mx = sv;
#pragma unroll
                    for (int m = 8; m >= 1; m >>= 1) mx = fmaxf(mx, __shfl_xor(mx, m, 32));
                    float e = __expf(sv - mx);
                    float sum = e;
#pragma unroll
                    for (int m = 8; m >= 1; m >>= 1) sum += __shfl_xor(sum, m, 32);
                    wp[(r + half * 8) * 16 + n] = e / sum;
                }
                float w2r[8];
#pragma unroll
                for (int r = 0; r < 8; ++r) {
                    int arow = r + half * 8;
                    float acc2 = 0.f;
#pragma unroll 1
                    for (int u = 0; u < 16; ++u)
                        acc2 += wp[arow * 16 + u] * v2w[h * 256 + u * 16 + n];
                    w2r[r] = acc2;
                }
#pragma unroll
                for (int r = 0; r < 8; ++r) wp[(r + half * 8) * 16 + n] = w2r[r];
                float hsv = hs[h];
#pragma unroll 1
                for (int i = 0; i < 32; ++i) {
                    int idx = i * 32 + lane;
                    int a2 = idx >> 7, d = idx & 127;
                    float o = 0.f;
#pragma unroll 1
                    for (int t2 = 0; t2 < 16; ++t2)
                        o += wp[a2 * 16 + t2] * bf2f(V1[b * 2048 + t2 * 128 + d]);
                    int tdest = s * 8 + a2;
                    xag[((size_t)(b0 + b) * T_ + tdest) * FFIN_ + h * 128 + d] =
                        f2bf(o * hsv);
                }
            }
        }  // heads

        // residual slice
        {
            float hres = hs[H_];
            for (int i = tid; i < BT_ * 8 * D_; i += NTHR) {
                int b = i >> 10, rem = i & 1023;
                int a2 = rem >> 7, d = rem & 127;
                int src = abase + a2, tdest = s * 8 + a2;
                float v = bf2f(XLN[(b * 16 + src) * 128 + d]) * hres;
                xag[((size_t)(b0 + b) * T_ + tdest) * FFIN_ + 1024 + d] = f2bf(v);
            }
        }
    }  // streams

    // make xag stores visible before TDM reads them back
    asm volatile("s_wait_storecnt 0x0" ::: "memory");
    __syncthreads();

    // ================= FFN (per-t weights) =================
    unsigned short* XAT   = (unsigned short*)(smem + OFF_XAT);
    float*          MID   = (float*)(smem + OFF_MID);
    unsigned short* HB    = (unsigned short*)(smem + OFF_HB);
    float*          OUTB  = (float*)(smem + OFF_OUTB);
    float*          FFOUT = (float*)(smem + OFF_FFOUT);
    const float* ff_in_b  = p.in[31];
    const float* ff_out_b = p.in[33];

#pragma unroll 1
    for (int t = 0; t < T_; ++t) {
        // TDM: DMA xa[:, t, :] tile (16 rows x 1152 bf16, row stride 18432) into XAT
        if (wave == 0) {
            unsigned long long ga =
                (unsigned long long)(size_t)(xag + ((size_t)b0 * T_ + t) * FFIN_);
            tdm_load_2d(ga, OFF_XAT, FFIN_, BT_, T_ * FFIN_, /*2B*/ 1);
            __builtin_amdgcn_s_wait_tensorcnt(0);
        }
        __syncthreads();

        const unsigned short* W1b = ws + WSE_FFINW + (size_t)t * 2 * DFF_ * FFIN_;
        // GEMM1: mid[16][1024] = xa[16][1152] * W1^T
#pragma unroll 1
        for (int nt = wave; nt < 64; nt += NWAVE) {
            v8f acc = zero8();
#pragma unroll 1
            for (int kt = 0; kt < 36; ++kt) {
                if (kt + 1 < 36)
                    __builtin_prefetch(W1b + (size_t)(nt * 16) * FFIN_ + (kt + 1) * 32, 0, 0);
                stage_b_tile_bf(stg, W1b, FFIN_, nt * 16, kt * 32, 2 * DFF_);
                v16bf af = load_a_frag(XAT, FFIN_, kt * 32);
                v16bf bf_ = load_b_frag(stg);
                acc = wmma_bf16(af, bf_, acc);
            }
            int n = lane & 15, half = lane >> 4;
            float bb = ff_in_b[(size_t)t * 2 * DFF_ + nt * 16 + n];
#pragma unroll
            for (int r = 0; r < 8; ++r)
                MID[(r + half * 8) * 1024 + nt * 16 + n] = acc[r] + bb;
        }
        __syncthreads();

        // silu gate -> HB bf16 [16][512]
        for (int i = tid; i < BT_ * DFF_; i += NTHR) {
            int b = i >> 9, j = i & 511;
            float m1 = MID[b * 1024 + j];
            float m2 = MID[b * 1024 + DFF_ + j];
            float hv = (m1 / (1.f + __expf(-m1))) * m2;
            HB[b * DFF_ + j] = f2bf(hv);
        }
        __syncthreads();

        const unsigned short* W2b = ws + WSE_FFOUTW + (size_t)t * DOUT_ * DFF_;
        // GEMM2: ff[16][129] = h[16][512] * W2^T
#pragma unroll 1
        for (int nt = wave; nt < 9; nt += NWAVE) {
            v8f acc = zero8();
#pragma unroll 1
            for (int kt = 0; kt < 16; ++kt) {
                stage_b_tile_bf(stg, W2b, DFF_, nt * 16, kt * 32, DOUT_);
                v16bf af = load_a_frag(HB, DFF_, kt * 32);
                v16bf bf_ = load_b_frag(stg);
                acc = wmma_bf16(af, bf_, acc);
            }
            int n = lane & 15, half = lane >> 4;
            int o = nt * 16 + n;
            if (o < DOUT_) {
                float bb = ff_out_b[(size_t)t * DOUT_ + o];
#pragma unroll
                for (int r = 0; r < 8; ++r)
                    FFOUT[(r + half * 8) * 144 + o] = acc[r] + bb;
            }
        }
        __syncthreads();

        // resid gate -> OUTB [16][128] f32 (HB dead now)
        for (int i = tid; i < BT_ * D_; i += NTHR) {
            int b = i >> 7, d = i & 127;
            float gate = (1.f / (1.f + __expf(-FFOUT[b * 144 + 128]))) * 2.0f;  // RGM
            OUTB[b * 128 + d] = FFOUT[b * 144 + d] * gate;
        }
        __syncthreads();

        // TDM: DMA OUTB (16 rows x 128 f32, dest row stride 1024 f32) to d_out
        if (wave == 0) {
            float* dst = (t < 8)
                ? (out + ((size_t)b0 * 8 + t) * D_)
                : (out + (size_t)B_ * 8 * D_ + ((size_t)b0 * 8 + (t - 8)) * D_);
            tdm_store_2d((unsigned long long)(size_t)dst, OFF_OUTB, D_, BT_,
                         8 * D_, /*4B*/ 2);
            __builtin_amdgcn_s_wait_tensorcnt(0);
        }
        __syncthreads();  // before reusing XAT/MID/OUTB next t
    }
}

extern "C" void kernel_launch(void* const* d_in, const int* in_sizes, int n_in,
                              void* d_out, int out_size, void* d_ws, size_t ws_size,
                              hipStream_t stream) {
    (void)in_sizes; (void)n_in; (void)out_size; (void)ws_size;
    Params p;
    for (int i = 0; i < 34; ++i) p.in[i] = (const float*)d_in[i];
    unsigned short* ws = (unsigned short*)d_ws;
    float* out = (float*)d_out;

    // once-per-launch bf16 weight conversion into d_ws (deterministic)
    struct { int idx; size_t wsoff; int n; } jobs[12] = {
        { 8, WSE_QW0,    8 * HD_ * D_ },   // ai_qw
        {19, WSE_QW1,    8 * HD_ * D_ },   // ae_qw
        {10, WSE_KAW0,   HD_ * D_ },       // ai_kaw
        {21, WSE_KAW1,   HD_ * D_ },       // ae_kaw
        {12, WSE_KBW0,   HD_ * D_ },       // ai_kbw
        {23, WSE_KBW1,   HD_ * D_ },       // ae_kbw
        {14, WSE_V1AW0,  D_ * D_ },        // ai_v1aw
        {25, WSE_V1AW1,  D_ * D_ },        // ae_v1aw
        {16, WSE_V1BW0,  D_ * D_ },        // ai_v1bw
        {27, WSE_V1BW1,  D_ * D_ },        // ae_v1bw
        {30, WSE_FFINW,  T_ * 2 * DFF_ * FFIN_ },  // ff_in_w
        {32, WSE_FFOUTW, T_ * DOUT_ * DFF_ },      // ff_out_w
    };
    for (int j = 0; j < 12; ++j) {
        int blocks = (jobs[j].n + NTHR * 8 - 1) / (NTHR * 8);
        cvt_bf16<<<blocks, NTHR, 0, stream>>>((const float*)d_in[jobs[j].idx],
                                              ws + jobs[j].wsoff, jobs[j].n);
    }

    (void)hipFuncSetAttribute(reinterpret_cast<const void*>(fused_layer),
                              hipFuncAttributeMaxDynamicSharedMemorySize, SMEM_BYTES);
    fused_layer<<<B_ / BT_, NTHR, SMEM_BYTES, stream>>>(p, ws, out);
}
